// MHSelfAttention_61967788146887
// MI455X (gfx1250) — matrix operand order repaired and checked by reading, not compiled, not run on hardware
//
#include <hip/hip_runtime.h>
#include <hip/hip_bf16.h>
#include <stdint.h>

// ---------------- problem constants ----------------
#define DIMV   1024
#define HEADS  16
#define HD     64
#define SEQ    2048
#define BATCH  4
#define EOUT   (3 * DIMV)          // 3072 qkv output columns
#define MROWS  (BATCH * SEQ)       // 8192 rows of x
#define PLANE  ((size_t)BATCH * HEADS * SEQ * HD)   // elements per Q/K/V tensor

typedef __attribute__((ext_vector_type(16))) _Float16 v16h;
typedef __attribute__((ext_vector_type(8)))  _Float16 v8h;
typedef __attribute__((ext_vector_type(8)))  float    v8f;
typedef __attribute__((ext_vector_type(4)))  float    v4f;
typedef __attribute__((ext_vector_type(4)))  unsigned int u32x4;
typedef __attribute__((ext_vector_type(8)))  int      i32x8;
typedef __attribute__((ext_vector_type(4)))  int      i32x4;

#if __has_builtin(__builtin_amdgcn_tensor_load_to_lds) && __has_builtin(__builtin_amdgcn_s_wait_tensorcnt)
#define HAVE_TDM 1
#else
#define HAVE_TDM 0
#endif

static __device__ __forceinline__ v8f wmma_f16(v16h a, v16h b, v8f c) {
    // D = A(16x32 f16) * B(32x16 f16) + C(16x16 f32)
    return __builtin_amdgcn_wmma_f32_16x16x32_f16(
        /*neg_a=*/false, a, /*neg_b=*/false, b,
        /*c_mod=*/(short)0, c, /*reuse_a=*/false, /*reuse_b=*/false);
}

// =====================================================================
// Kernel 1: qkv = x @ W^T  (f32 in, f16 out), scattered to [3][B,H,N,hd]
//   block = 256 threads (8 waves). Block tile: 128 (m) x 128 (e).
//   wave tile: 32x64 = 2x4 WMMA fragments (8 wmma / k-step).
// =====================================================================
__global__ __launch_bounds__(256)
void qkv_gemm_kernel(const float* __restrict__ x,
                     const float* __restrict__ W,
                     _Float16* __restrict__ qkv)
{
    __shared__ __align__(32) _Float16 Xs[128 * 32];
    __shared__ __align__(32) _Float16 Ws[128 * 32];

    const int tid  = threadIdx.x;
    const int lane = tid & 31;
    const int wave = tid >> 5;          // 0..7
    const int wrow = wave >> 1;         // 0..3 : 32-row strip (m)
    const int wcol = wave & 1;          // 0..1 : 64-col strip (e)
    const int m0   = blockIdx.x * 128;
    const int e0   = blockIdx.y * 128;

    const int ln16 = lane & 15;
    const int half = lane >> 4;

    v8f acc[2][4];
    for (int i = 0; i < 2; ++i)
        for (int j = 0; j < 4; ++j)
            for (int r = 0; r < 8; ++r) acc[i][j][r] = 0.0f;

    for (int k0 = 0; k0 < DIMV; k0 += 32) {
        __syncthreads();
        // ---- stage X tile 128x32 (f32 -> f16) : 1024 vec4 slots
        for (int i = 0; i < 4; ++i) {
            int idx = tid + i * 256;            // 0..1023
            int row = idx >> 3;                 // 8 vec4 per 32-wide row
            int c4  = (idx & 7) * 4;
            v4f xv = *(const v4f*)(x + (size_t)(m0 + row) * DIMV + k0 + c4);
            _Float16* d = &Xs[row * 32 + c4];
            d[0] = (_Float16)xv.x; d[1] = (_Float16)xv.y;
            d[2] = (_Float16)xv.z; d[3] = (_Float16)xv.w;
        }
        // ---- stage W tile 128x32 : 1024 vec4 slots
        for (int i = 0; i < 4; ++i) {
            int idx = tid + i * 256;
            int row = idx >> 3;
            int c4  = (idx & 7) * 4;
            v4f wv = *(const v4f*)(W + (size_t)(e0 + row) * DIMV + k0 + c4);
            _Float16* d = &Ws[row * 32 + c4];
            d[0] = (_Float16)wv.x; d[1] = (_Float16)wv.y;
            d[2] = (_Float16)wv.z; d[3] = (_Float16)wv.w;
        }
        __syncthreads();

        // ---- A fragments (rows of X), documented 16-bit A layout
        v16h afrag[2];
        const int kbA = half * 8;
        for (int mi = 0; mi < 2; ++mi) {
            const _Float16* p = &Xs[(wrow * 32 + mi * 16 + ln16) * 32];
            v8h lo = *(const v8h*)(p + kbA);
            v8h hi = *(const v8h*)(p + kbA + 16);
            for (int j = 0; j < 8; ++j) { afrag[mi][j] = lo[j]; afrag[mi][8 + j] = hi[j]; }
        }
        // ---- B fragments: B(k,n) = W[e0+..+n][k0+k]; lane = n, table k order
        v16h bfrag[4];
        const int kbB = half * 8;
        for (int ni = 0; ni < 4; ++ni) {
            const _Float16* p = &Ws[(wcol * 64 + ni * 16 + ln16) * 32 + kbB];
            v8h lo = *(const v8h*)(p);
            v8h hi = *(const v8h*)(p + 16);
            for (int j = 0; j < 8; ++j) { bfrag[ni][j] = lo[j]; bfrag[ni][8 + j] = hi[j]; }
        }
        for (int mi = 0; mi < 2; ++mi)
            for (int ni = 0; ni < 4; ++ni)
                acc[mi][ni] = wmma_f16(afrag[mi], bfrag[ni], acc[mi][ni]);
    }

    // ---- scatter to [which][B,H,N,hd] as f16
    for (int mi = 0; mi < 2; ++mi) {
        for (int ni = 0; ni < 4; ++ni) {
            int e      = e0 + wcol * 64 + ni * 16 + ln16;
            int which  = e >> 10;          // 0=Q 1=K 2=V
            int dinner = e & 1023;
            int h      = dinner >> 6;
            int dd     = dinner & 63;
            for (int r = 0; r < 8; ++r) {
                int m = m0 + wrow * 32 + mi * 16 + r + 8 * half;
                int b = m >> 11;
                int n = m & 2047;
                size_t off = (size_t)which * PLANE
                           + ((size_t)(b * HEADS + h) * SEQ + n) * HD + dd;
                qkv[off] = (_Float16)acc[mi][ni][r];
            }
        }
    }
}

// =====================================================================
// Kernel 2: flash attention.  block = 256 threads (8 waves).
//   grid = (SEQ/128, B*H). Wave w owns query rows [q0, q0+16).
//   K tile staged by the Tensor Data Mover (TENSORcnt); V staged
//   transposed cooperatively. 8 WMMAs per 32-key tile per wave.
// =====================================================================
__global__ __launch_bounds__(256)
void attn_kernel(const _Float16* __restrict__ qkv, float* __restrict__ out)
{
    __shared__ __align__(32) _Float16 Ks[32 * 64];       // [key][d]
    __shared__ __align__(32) _Float16 VsT[64 * 32];      // [d][key]
    __shared__ __align__(32) _Float16 Ps[8][16 * 32];    // per-wave P tile

    const int tid  = threadIdx.x;
    const int lane = tid & 31;
    const int wave = tid >> 5;
    const int ln16 = lane & 15;
    const int half = lane >> 4;

    const int bh = blockIdx.y;              // 0..63
    const int b  = bh >> 4;
    const int h  = bh & 15;
    const int q0 = blockIdx.x * 128 + wave * 16;

    const size_t planeOff = (size_t)bh * SEQ * HD;
    const _Float16* Qg = qkv + planeOff;
    const _Float16* Kg = qkv + PLANE + planeOff;
    const _Float16* Vg = qkv + 2 * PLANE + planeOff;

#if HAVE_TDM
    const unsigned ksLds = (unsigned)(uintptr_t)(void*)&Ks[0];  // LDS byte offset (flat addr low 32)
#endif

    // ---- Q A-fragments for d 0..31 and 32..63 (resident whole kernel)
    v16h qf[2];
    {
        const _Float16* qrow = Qg + (size_t)(q0 + ln16) * HD;
        const int kb = half * 8;
        for (int s = 0; s < 2; ++s) {
            v8h lo = *(const v8h*)(qrow + s * 32 + kb);
            v8h hi = *(const v8h*)(qrow + s * 32 + kb + 16);
            for (int j = 0; j < 8; ++j) { qf[s][j] = lo[j]; qf[s][8 + j] = hi[j]; }
        }
    }

    v8f o[4];
    for (int dt = 0; dt < 4; ++dt)
        for (int r = 0; r < 8; ++r) o[dt][r] = 0.0f;

    float m_r[8], l_r[8];
    for (int r = 0; r < 8; ++r) { m_r[r] = -1.0e30f; l_r[r] = 0.0f; }

    const float scale = 0.125f;             // hd^-0.5, hd = 64

    for (int j0 = 0; j0 < SEQ; j0 += 32) {
        __syncthreads();                    // prior iteration done with Ks/VsT
#if HAVE_TDM
        if (wave == 0) {
            // Tensor DMA: 2D tile [32 keys x 64 d] f16, row-major -> Ks
            uint64_t gaddr = (uint64_t)(uintptr_t)(Kg + (size_t)j0 * HD);
            u32x4 g0;
            g0[0] = 1u;                                    // count=1, user mode
            g0[1] = ksLds;                                 // lds_addr
            g0[2] = (unsigned)(gaddr & 0xFFFFFFFFu);       // global_addr[31:0]
            g0[3] = (unsigned)((gaddr >> 32) & 0x1FFFFFFu) // global_addr[56:32]
                  | (2u << 30);                            // type = 2 ("image")
            i32x8 g1;
            g1[0] = (int)(1u << 16);            // wg_mask=0, data_size=1 (2 bytes)
            g1[1] = (int)(64u << 16);           // tensor_dim0 = 64 (bits 79:48)
            g1[2] = (int)((unsigned)SEQ << 16); // tensor_dim1 = 2048 (bits 111:80)
            g1[3] = (int)(64u << 16);           // tile_dim0 = 64 (bits 127:112)
            g1[4] = 32;                         // tile_dim1 = 32 (bits 143:128)
            g1[5] = 64;                         // tensor_dim0_stride = 64 (bits 207:160)
            g1[6] = 0;
            g1[7] = 0;
            i32x4 z4;
            z4[0] = 0; z4[1] = 0; z4[2] = 0; z4[3] = 0;
            i32x8 z8;
            for (int j = 0; j < 8; ++j) z8[j] = 0;
            __builtin_amdgcn_tensor_load_to_lds(g0, g1, z4, z4, z8, 0);
            __builtin_amdgcn_s_wait_tensorcnt(0);
        }
#else
        {
            int row = tid >> 3;             // key 0..31
            int c8  = (tid & 7) * 8;        // d offset
            *(v8h*)(&Ks[row * 64 + c8]) =
                *(const v8h*)(Kg + (size_t)(j0 + row) * HD + c8);
        }
#endif
        // ---- cooperative transposed stage of V (for contiguous B-frags)
        {
            int row = tid >> 3;             // key 0..31
            int c8  = (tid & 7) * 8;        // d offset
            v8h vv = *(const v8h*)(Vg + (size_t)(j0 + row) * HD + c8);
            for (int j = 0; j < 8; ++j) VsT[(c8 + j) * 32 + row] = vv[j];
        }
        __syncthreads();

        // ---- S = Q K^T : two 16-key column tiles, contraction hd=64 (2 steps)
        v8f sfrag[2];
        for (int jj = 0; jj < 2; ++jj) {
            v8f s;
            for (int r = 0; r < 8; ++r) s[r] = 0.0f;
            for (int ds = 0; ds < 2; ++ds) {
                const _Float16* kp = &Ks[(jj * 16 + ln16) * 64 + ds * 32 + half * 8];
                v8h lo = *(const v8h*)(kp);
                v8h hi = *(const v8h*)(kp + 16);
                v16h bf;
                for (int j = 0; j < 8; ++j) { bf[j] = lo[j]; bf[8 + j] = hi[j]; }
                s = wmma_f16(qf[ds], bf, s);
            }
            sfrag[jj] = s;
        }

        // ---- online softmax, fully in registers.
        // Row (r + 8*half) lives across the 16 lanes of this half;
        // xor-shuffles over bits 0..3 never cross the half boundary.
        float p0[8], p1[8], alpha[8];
        for (int r = 0; r < 8; ++r) {
            float s0 = sfrag[0][r] * scale;
            float s1 = sfrag[1][r] * scale;
            float mx = fmaxf(s0, s1);
            for (int off = 8; off >= 1; off >>= 1)
                mx = fmaxf(mx, __shfl_xor(mx, off, 32));
            float mn = fmaxf(m_r[r], mx);
            alpha[r] = __expf(m_r[r] - mn);
            p0[r] = __expf(s0 - mn);
            p1[r] = __expf(s1 - mn);
            float ts = p0[r] + p1[r];
            for (int off = 8; off >= 1; off >>= 1)
                ts += __shfl_xor(ts, off, 32);
            l_r[r] = alpha[r] * l_r[r] + ts;
            m_r[r] = mn;
        }
        for (int dt = 0; dt < 4; ++dt)
            for (int r = 0; r < 8; ++r) o[dt][r] *= alpha[r];

        // ---- P: C-layout registers -> LDS -> A-layout fragment (f16)
        _Float16* pp = &Ps[wave][0];
        for (int r = 0; r < 8; ++r) {
            int row = r + 8 * half;
            pp[row * 32 + ln16]      = (_Float16)p0[r];
            pp[row * 32 + 16 + ln16] = (_Float16)p1[r];
        }
        v16h pfrag;
        {
            const _Float16* prow = pp + ln16 * 32;
            const int kb = half * 8;
            v8h lo = *(const v8h*)(prow + kb);
            v8h hi = *(const v8h*)(prow + kb + 16);
            for (int j = 0; j < 8; ++j) { pfrag[j] = lo[j]; pfrag[8 + j] = hi[j]; }
        }

        // ---- O += P * V : contraction over 32 keys, 4 d-tiles of 16
        for (int dt = 0; dt < 4; ++dt) {
            const _Float16* vp = &VsT[(dt * 16 + ln16) * 32 + half * 8];
            v8h lo = *(const v8h*)(vp);
            v8h hi = *(const v8h*)(vp + 16);
            v16h vf;
            for (int j = 0; j < 8; ++j) { vf[j] = lo[j]; vf[8 + j] = hi[j]; }
            o[dt] = wmma_f16(pfrag, vf, o[dt]);
        }
    }

    // ---- final normalize + store  out[b][n][h*64+d], f32
    float inv_l[8];
    for (int r = 0; r < 8; ++r) inv_l[r] = 1.0f / l_r[r];
    for (int dt = 0; dt < 4; ++dt) {
        for (int r = 0; r < 8; ++r) {
            int qrow = q0 + r + 8 * half;
            int d    = dt * 16 + ln16;
            out[((size_t)(b * SEQ + qrow)) * DIMV + h * HD + d] = o[dt][r] * inv_l[r];
        }
    }
}

// =====================================================================
extern "C" void kernel_launch(void* const* d_in, const int* in_sizes, int n_in,
                              void* d_out, int out_size, void* d_ws, size_t ws_size,
                              hipStream_t stream) {
    const float* x = (const float*)d_in[0];   // [4,2048,1024] f32
    const float* W = (const float*)d_in[1];   // [3072,1024]  f32
    float* out = (float*)d_out;               // [4,2048,1024] f32
    _Float16* qkvws = (_Float16*)d_ws;        // 3 * 8M f16 = 48 MB

    // QKV projection: grid (M/128, E/128) = (64, 24)
    qkv_gemm_kernel<<<dim3(MROWS / 128, EOUT / 128), 256, 0, stream>>>(x, W, qkvws);
    // Flash attention: grid (SEQ/128, B*H) = (16, 64)
    attn_kernel<<<dim3(SEQ / 128, BATCH * HEADS), 256, 0, stream>>>(qkvws, out);
}
